// SwinTransformerBlock_KA_32590211842613
// MI455X (gfx1250) — matrix operand order repaired and checked by reading, not compiled, not run on hardware
//
#include <hip/hip_runtime.h>
#include <math.h>

// ---------------------------------------------------------------------------
// Problem constants (B=4, R=64, C=192, WN=16, K=3, WS=16)
// ---------------------------------------------------------------------------
#define BATCH   4
#define RES     64
#define CH      192
#define NWIN    16
#define NPIX    4096          // RES*RES
#define KTOT    1728          // CH * 9 taps
#define KSTEPS  54            // KTOT / 32
#define NTILES  12            // CH / 16

typedef __attribute__((ext_vector_type(16))) __bf16 v16bf;
typedef __attribute__((ext_vector_type(8)))  float  v8f;

union FragAB { v16bf v; uint4 q[2]; };

__device__ __forceinline__ unsigned short f2bf(float f) {
  union { float f; unsigned int u; } v; v.f = f;
  unsigned int r = v.u + 0x7FFFu + ((v.u >> 16) & 1u);   // round-to-nearest-even
  return (unsigned short)(r >> 16);
}

// ---------------------------------------------------------------------------
// Kernel X: x (f32) -> xbf (bf16), 8 elements per thread, fully vectorized.
// n = 4*4096*192 = 3,145,728 -> 393216 threads (exact, no tail).
// ---------------------------------------------------------------------------
__global__ void k_cvt_bf16(const float* __restrict__ x,
                           unsigned short* __restrict__ xbf) {
  size_t i = (size_t)blockIdx.x * blockDim.x + threadIdx.x;   // per 8 floats
  const float4* s = (const float4*)x + i * 2;
  float4 f0 = s[0], f1 = s[1];
  union { unsigned short u[8]; uint4 q; } p;
  p.u[0] = f2bf(f0.x); p.u[1] = f2bf(f0.y); p.u[2] = f2bf(f0.z); p.u[3] = f2bf(f0.w);
  p.u[4] = f2bf(f1.x); p.u[5] = f2bf(f1.y); p.u[6] = f2bf(f1.z); p.u[7] = f2bf(f1.w);
  ((uint4*)xbf)[i] = p.q;
}

// ---------------------------------------------------------------------------
// Kernel A: pooled[b,win,c] = mean over 16x16 of depthwise 3x3 conv (pad 1).
// Separable trick: pool(conv) = sum_t w[t] * S(di,dj) where S is a
// border-trimmed window sum -> 256 loads/thread instead of 2304.
// Grid: (B*NWIN) blocks x CH threads; loads coalesced over c.
// ---------------------------------------------------------------------------
__global__ void k_pool(const float* __restrict__ x,
                       const float* __restrict__ conv1_w,
                       float* __restrict__ pooled) {
  int blk = blockIdx.x;                  // b*16 + win
  int b = blk >> 4, win = blk & 15;
  int c = threadIdx.x;                   // 0..191
  int wy = win >> 2, wx = win & 3;

  const float* xb = x + (size_t)b * NPIX * CH + c;
  float SB[3] = {0.f, 0.f, 0.f};         // sum over all rows of R(u,dj)
  float R0[3], R15[3];
  for (int u = 0; u < 16; ++u) {
    const float* xr = xb + (size_t)((wy * 16 + u) * RES + wx * 16) * CH;
    float full = 0.f, first = 0.f, last = 0.f;
    for (int v = 0; v < 16; ++v) {
      float val = xr[(size_t)v * CH];
      full += val;
      if (v == 0)  first = val;
      if (v == 15) last  = val;
    }
    float r[3] = {full - last, full, full - first};   // dj = -1, 0, +1
    if (u == 0)  { R0[0] = r[0]; R0[1] = r[1]; R0[2] = r[2]; }
    if (u == 15) { R15[0] = r[0]; R15[1] = r[1]; R15[2] = r[2]; }
    SB[0] += r[0]; SB[1] += r[1]; SB[2] += r[2];
  }
  const float* kw = conv1_w + (size_t)(win * CH + c) * 9;
  float acc = 0.f;
  for (int ki = 0; ki < 3; ++ki)
    for (int kj = 0; kj < 3; ++kj) {
      float S = SB[kj];
      if (ki == 0) S -= R15[kj];     // di = -1: exclude u=15
      if (ki == 2) S -= R0[kj];      // di = +1: exclude u=0
      acc += kw[ki * 3 + kj] * S;
    }
  pooled[(size_t)(b * NWIN + win) * CH + c] = acc * (1.0f / 256.0f);
}

// ---------------------------------------------------------------------------
// Kernel B: per-batch reweight MLP. Grid: B blocks x 64 threads.
//   w0 = pooled . dc_w + dc_b ; h = gelu(w0 @ l1^T + b1) (exact erf)
//   w  = sigmoid(h @ l2^T + b2) ; also gw = gk_w * w
// ---------------------------------------------------------------------------
__global__ void k_mlp(const float* __restrict__ pooled,
                      const float* __restrict__ dc_w, const float* __restrict__ dc_b,
                      const float* __restrict__ l1_w, const float* __restrict__ l1_b,
                      const float* __restrict__ l2_w, const float* __restrict__ l2_b,
                      const float* __restrict__ gk_w,
                      float* __restrict__ wwin, float* __restrict__ gw) {
  int b = blockIdx.x, tid = threadIdx.x;   // 64 threads
  __shared__ float s_w0[NWIN], s_h[64];
  if (tid < NWIN) {
    float acc = dc_b[tid];
    const float* p = pooled + (size_t)(b * NWIN + tid) * CH;
    for (int c = 0; c < CH; ++c) acc += p[c] * dc_w[tid * CH + c];
    s_w0[tid] = acc;
  }
  __syncthreads();
  {
    float acc = l1_b[tid];
    for (int wn = 0; wn < NWIN; ++wn) acc += s_w0[wn] * l1_w[tid * NWIN + wn];
    s_h[tid] = 0.5f * acc * (1.0f + erff(acc * 0.70710678118654752f));
  }
  __syncthreads();
  if (tid < NWIN) {
    float acc = l2_b[tid];
    for (int j = 0; j < 64; ++j) acc += s_h[j] * l2_w[tid * 64 + j];
    float sg = 1.0f / (1.0f + expf(-acc));
    wwin[b * NWIN + tid] = sg;
    gw[b * NWIN + tid] = sg * gk_w[tid];
  }
}

// ---------------------------------------------------------------------------
// Kernel C: build Keff in exact WMMA B-fragment lane order (bf16).
// Keff[b,co,ci,t] = sum_win fw[co,win*C+ci]*w[b,win]*k1[win*C+ci,t]
//                 + fw[co,16C+ci]*(sum_win gw[b,win]*k1[win*C+ci,t] + gk_b)
// Blob layout: [b][s][nt][lane][half] so kernel D lanes read 32B contiguous.
// Grid: B*KSTEPS*NTILES = 2592 blocks x 512 threads (= lane*16 + half).
// ---------------------------------------------------------------------------
__global__ void k_keff(const float* __restrict__ conv1_w,
                       const float* __restrict__ fusion_w,
                       const float* __restrict__ wwin,
                       const float* __restrict__ gw,
                       const float* __restrict__ gk_b,
                       unsigned short* __restrict__ keff) {
  int blk = blockIdx.x;                 // ((b*54 + s)*12 + nt)
  int nt = blk % NTILES;
  int bs = blk / NTILES;
  int s = bs % KSTEPS;
  int b = bs / KSTEPS;
  int tid = threadIdx.x;                // 0..511
  int half = tid & 15, lane = tid >> 4; // lane 0..31
  int klocal = ((half >> 3) << 4) + ((lane >> 4) << 3) + (half & 7);      // 0..31 within the K-step
  int t  = s / 6;
  int ci = (s % 6) * 32 + klocal;
  int co = nt * 16 + (lane & 15);

  const float* fw = fusion_w + (size_t)co * ((NWIN + 1) * CH);
  const float* ww = wwin + b * NWIN;
  const float* g  = gw + b * NWIN;
  float acc = 0.f, gacc = 0.f;
  for (int wn = 0; wn < NWIN; ++wn) {
    float k1 = conv1_w[(size_t)(wn * CH + ci) * 9 + t];
    acc  += fw[wn * CH + ci] * ww[wn] * k1;
    gacc += g[wn] * k1;
  }
  acc += fw[NWIN * CH + ci] * (gacc + gk_b[0]);
  keff[(size_t)blk * 512 + tid] = f2bf(acc);
}

// ---------------------------------------------------------------------------
// Kernel D: fused 3x3 conv as im2col GEMM, M=4096 N=192 K=1728 per batch.
// v_wmma_f32_16x16x32_bf16, operands streamed from L2 with b128 loads
// (x bf16 6.3MB + Keff 2.6MB are L2-resident on the 192MB L2).
// Grid: 256 blocks (b, image row) x 256 threads (8 waves).
// Wave w: M-subtile mt = w>>1 (16 pixels of the 64-pixel row),
//         N-tiles nt = (w&1)*6 .. +5  -> 6 f32 accumulators (48 VGPRs).
// Border zero-padding via predicated A loads; EXEC all-1s at every WMMA.
// ---------------------------------------------------------------------------
__global__ void k_conv_wmma(const unsigned short* __restrict__ xbf,
                            const unsigned short* __restrict__ keff,
                            const float* __restrict__ fusion_b,
                            float* __restrict__ out) {
  int blk = blockIdx.x;                 // b*64 + rowblk
  int b = blk >> 6, rowblk = blk & 63;
  int tid = threadIdx.x;
  int wave = tid >> 5, lane = tid & 31;
  int mt = wave >> 1;                   // 0..3
  int ntbase = (wave & 1) * 6;          // 0 or 6
  int m = lane & 15, hi = lane >> 4;

  v8f acc[6] = {};                      // zero-init accumulators

  const unsigned short* keffb = keff + (size_t)b * KSTEPS * NTILES * 512;

  for (int t = 0; t < 9; ++t) {
    int di = t / 3 - 1, dj = t % 3 - 1;
    int hp = rowblk + di;
    int wp = mt * 16 + m + dj;
    bool valid = (hp >= 0) & (hp < RES) & (wp >= 0) & (wp < RES);
    // A row base for this lane's pixel (only dereferenced when valid)
    long aoff = ((long)(b * NPIX + hp * RES + wp)) * CH + hi * 8;

    for (int kc = 0; kc < 6; ++kc) {
      FragAB a;
      a.q[0] = uint4{0, 0, 0, 0};
      a.q[1] = uint4{0, 0, 0, 0};
      if (valid) {
        const unsigned short* ap = xbf + aoff + kc * 32;
        a.q[0] = *(const uint4*)(ap);        // halves 0..7 : k = hi*8 + 0..7
        a.q[1] = *(const uint4*)(ap + 16);   // halves 8..15: k = 16+hi*8+0..7
      }
      int s = t * 6 + kc;
      const unsigned short* bb =
          keffb + ((size_t)(s * NTILES + ntbase)) * 512 + lane * 16;
      if (s < KSTEPS - 1)                     // prefetch next K-step's B data
        __builtin_prefetch(bb + NTILES * 512, 0, 0);
#pragma unroll
      for (int j = 0; j < 6; ++j) {
        FragAB bf;
        const uint4* bp = (const uint4*)(bb + j * 512);
        bf.q[0] = bp[0];
        bf.q[1] = bp[1];
        acc[j] = __builtin_amdgcn_wmma_f32_16x16x32_bf16(
            false, a.v, false, bf.v, (short)0, acc[j], false, false);
      }
    }
  }

  // Epilogue: D layout row = hi*8 + i, col = lane&15. Add fusion bias.
  float* ob = out + ((size_t)(b * NPIX + rowblk * RES)) * CH;
#pragma unroll
  for (int j = 0; j < 6; ++j) {
    int co = (ntbase + j) * 16 + m;
    float bias = fusion_b[co];
    int wbase = mt * 16 + hi * 8;
#pragma unroll
    for (int i = 0; i < 8; ++i) {
      ob[(size_t)(wbase + i) * CH + co] = acc[j][i] + bias;
    }
  }
}

// ---------------------------------------------------------------------------
// Launcher
// ---------------------------------------------------------------------------
extern "C" void kernel_launch(void* const* d_in, const int* in_sizes, int n_in,
                              void* d_out, int out_size, void* d_ws, size_t ws_size,
                              hipStream_t stream) {
  const float* x        = (const float*)d_in[0];
  const float* conv1_w  = (const float*)d_in[1];
  const float* dc_w     = (const float*)d_in[2];
  const float* dc_b     = (const float*)d_in[3];
  const float* l1_w     = (const float*)d_in[4];
  const float* l1_b     = (const float*)d_in[5];
  const float* l2_w     = (const float*)d_in[6];
  const float* l2_b     = (const float*)d_in[7];
  const float* gk_w     = (const float*)d_in[8];
  const float* gk_b     = (const float*)d_in[9];
  const float* fusion_w = (const float*)d_in[10];
  const float* fusion_b = (const float*)d_in[11];
  float* out = (float*)d_out;

  char* ws = (char*)d_ws;
  float*          pooled = (float*)(ws + 0);            //  49152 B
  float*          wwin   = (float*)(ws + 49152);        //    256 B
  float*          gw     = (float*)(ws + 49664);        //    256 B
  unsigned short* xbf    = (unsigned short*)(ws + 65536);     // 6,291,456 B
  unsigned short* keff   = (unsigned short*)(ws + 6356992);   // 2,654,208 B

  // 1) x -> bf16 (3,145,728 elems / 8 per thread = 393216 threads, exact)
  k_cvt_bf16<<<1536, 256, 0, stream>>>(x, xbf);
  // 2) pooled via separable window sums
  k_pool<<<BATCH * NWIN, CH, 0, stream>>>(x, conv1_w, pooled);
  // 3) tiny per-batch MLP -> window weights
  k_mlp<<<BATCH, 64, 0, stream>>>(pooled, dc_w, dc_b, l1_w, l1_b,
                                  l2_w, l2_b, gk_w, wwin, gw);
  // 4) effective 3x3 kernel, pre-swizzled into WMMA B-fragment order
  k_keff<<<BATCH * KSTEPS * NTILES, 512, 0, stream>>>(conv1_w, fusion_w,
                                                      wwin, gw, gk_b, keff);
  // 5) fused conv-as-GEMM with bf16 WMMA + bias epilogue
  k_conv_wmma<<<BATCH * RES, 256, 0, stream>>>(xbf, keff, fusion_b, out);
}
